// RQTransformer_61177514165030
// MI455X (gfx1250) — compile-verified, matrix operand order checked by reading, not run on hardware
//
#include <hip/hip_runtime.h>
#include <hip/hip_bf16.h>
#include <cstdint>

// ---------------------------------------------------------------------------
// RQ-Transformer forward for MI455X (gfx1250, wave32, WMMA).
// GEMMs: v_wmma_f32_16x16x32_f16 with b128 LDS fragment loads;
// weights pre-packed K-chunk-major so every LDS/global access is 16B vector.
// LDS staging uses GLOBAL_LOAD_ASYNC_TO_LDS_B128 (ASYNCcnt) when the
// builtin is available, with a synchronous fallback otherwise.
// ---------------------------------------------------------------------------

typedef __attribute__((ext_vector_type(16))) _Float16 v16h;
typedef __attribute__((ext_vector_type(8)))  float    v8f;

union FragU { v16h v; uint4 q[2]; };

// ---- optional CDNA5 async-to-LDS path (compile-safe feature probe) --------
#ifndef RQ_HAVE_ASYNC
#if defined(__AMDGCN__) && defined(__has_builtin)
#if __has_builtin(__builtin_amdgcn_global_load_async_to_lds_b128)
#define RQ_HAVE_ASYNC 1
#endif
#endif
#endif
#ifndef RQ_HAVE_ASYNC
#define RQ_HAVE_ASYNC 0
#endif

#if RQ_HAVE_ASYNC
typedef int v4i_gcc __attribute__((vector_size(16)));
using gv4i = __attribute__((address_space(1))) v4i_gcc;  // global (AS1) int4
using lv4i = __attribute__((address_space(3))) v4i_gcc;  // LDS    (AS3) int4
__device__ __forceinline__ void async_b128(const void* g, void* l) {
  __builtin_amdgcn_global_load_async_to_lds_b128((gv4i*)g, (lv4i*)l, 0, 0);
}
__device__ __forceinline__ void wait_async0() {
#if __has_builtin(__builtin_amdgcn_s_wait_asynccnt)
  __builtin_amdgcn_s_wait_asynccnt(0);
#else
  asm volatile("s_wait_asynccnt 0" ::: "memory");
#endif
}
#endif

#define DIMF   512
#define HEADS  8
#define DHEAD  64
#define INNER  512
#define VOCAB  16384
#define DEPTHT 4
#define MAXS   512
#define BATCH  2
#define NSP    513              // spatial sequence length (start token + 512)
#define MSP    (BATCH * NSP)    // 1026 spatial rows
#define NDP    (DEPTHT + 1)     // 5
#define MDP    (MSP * NDP)      // 5130 depth rows

enum { FLAG_GELU = 1, FLAG_REMAP = 2 };

// ---------------------------------------------------------------------------
__device__ __forceinline__ float gelu_exact(float x) {
  return 0.5f * x * (1.0f + erff(x * 0.70710678118654752f));
}

// ---------------------------------------------------------------------------
__global__ void cvt_f32_to_f16(const float* __restrict__ s,
                               _Float16* __restrict__ d, size_t n) {
  size_t i  = (size_t)blockIdx.x * blockDim.x + threadIdx.x;
  size_t st = (size_t)gridDim.x * blockDim.x;
  for (; i < n; i += st) d[i] = (_Float16)s[i];
}

// ---------------------------------------------------------------------------
// Pack weight W[K,N] (f32, row-major) into f16 K-chunk-major:
// P[((k>>5)*N + n)*32 + (k&31)] = W[k*N + n].  Layer-stacked tensors fold
// transparently since K % 32 == 0.
__global__ void pack_w(const float* __restrict__ W, _Float16* __restrict__ P,
                       size_t KN, int N) {
  size_t i  = (size_t)blockIdx.x * blockDim.x + threadIdx.x;
  size_t st = (size_t)gridDim.x * blockDim.x;
  for (; i < KN; i += st) {
    size_t k = i / (size_t)N;
    int    n = (int)(i - k * (size_t)N);
    P[(((k >> 5) * (size_t)N + n) << 5) + (k & 31)] = (_Float16)W[i];
  }
}

// ---------------------------------------------------------------------------
// Token embedding: tokens[b,s,d,:] = tok_emb[ids]+dp_pos[d];
// xsp[b,0,:] = start_tok; xsp[b,s+1,:] = sum_d tokens + sp_pos[s].
__global__ __launch_bounds__(256) void embed_kernel(
    const int* __restrict__ ids, const float* __restrict__ tok_emb,
    const float* __restrict__ start_tok, const float* __restrict__ sp_pos,
    const float* __restrict__ dp_pos, float* __restrict__ tokens,
    float* __restrict__ xsp) {
  const int sx = blockIdx.x;   // 0..512
  const int b  = blockIdx.y;
  if (sx == 0) {
    for (int c = threadIdx.x; c < DIMF; c += 256)
      xsp[(size_t)(b * NSP) * DIMF + c] = start_tok[c];
    return;
  }
  const int s = sx - 1;
  int id[DEPTHT];
#pragma unroll
  for (int d = 0; d < DEPTHT; ++d) id[d] = ids[b * (MAXS * DEPTHT) + s * DEPTHT + d];
  for (int c = threadIdx.x; c < DIMF; c += 256) {
    float sum = 0.f;
#pragma unroll
    for (int d = 0; d < DEPTHT; ++d) {
      float t = tok_emb[(size_t)id[d] * DIMF + c] + dp_pos[d * DIMF + c];
      tokens[(((size_t)(b * MAXS + s) * DEPTHT) + d) * DIMF + c] = t;
      sum += t;
    }
    xsp[((size_t)(b * NSP + s + 1)) * DIMF + c] = sum + sp_pos[(size_t)s * DIMF + c];
  }
}

// ---------------------------------------------------------------------------
// LayerNorm over 512 features; optional f16 and/or f32 outputs.
__global__ __launch_bounds__(128) void ln512(
    const float* __restrict__ x, const float* __restrict__ g,
    const float* __restrict__ b, _Float16* __restrict__ outh,
    float* __restrict__ outf) {
  __shared__ float red[128];
  const int row = blockIdx.x;
  const int tid = threadIdx.x;
  const float* xr = x + (size_t)row * DIMF;
  float s = 0.f;
  for (int c = tid; c < DIMF; c += 128) s += xr[c];
  red[tid] = s; __syncthreads();
  for (int o = 64; o > 0; o >>= 1) { if (tid < o) red[tid] += red[tid + o]; __syncthreads(); }
  const float mean = red[0] * (1.0f / DIMF);
  __syncthreads();
  float vs = 0.f;
  for (int c = tid; c < DIMF; c += 128) { float d = xr[c] - mean; vs += d * d; }
  red[tid] = vs; __syncthreads();
  for (int o = 64; o > 0; o >>= 1) { if (tid < o) red[tid] += red[tid + o]; __syncthreads(); }
  const float inv = rsqrtf(red[0] * (1.0f / DIMF) + 1e-5f);
  for (int c = tid; c < DIMF; c += 128) {
    float y = (xr[c] - mean) * inv * g[c] + b[c];
    if (outh) outh[(size_t)row * DIMF + c] = (_Float16)y;
    if (outf) outf[(size_t)row * DIMF + c] = y;
  }
}

// ---------------------------------------------------------------------------
// WMMA GEMM: C[M,N] = A[M,K](f16 row-major) * B (f16 K-chunk packed)
//            (+bias)(+GELU)(+residual).
// 256 threads = 8 waves; block tile 64x128; wave computes 16x64 (4 accs).
// Every fragment = 2x ds_load_b128 (ISA 7.12.2: lane's 16 halves are uints
// {0..3,8..11} (lanes 0-15) or {4..7,12..15} (lanes 16-31) of a 64B K-row).
// FLAG_REMAP: A row r reads source row (r/2048)*2565 + r%2048 + 1 (logits).
__global__ __launch_bounds__(256) void gemm_wmma_f16(
    const _Float16* __restrict__ A, const _Float16* __restrict__ Bp,
    const float* __restrict__ bias, const float* __restrict__ residual,
    float* __restrict__ Cf, _Float16* __restrict__ Ch,
    int M, int N, int K, int flags) {
  // 56-half (112B) row stride: 16B-aligned, 28*r mod 64 is conflict-free.
  __shared__ __attribute__((aligned(16))) _Float16 As[64][56];
  __shared__ __attribute__((aligned(16))) _Float16 Bs[128][56];
  const int tid  = threadIdx.x;
  const int lane = tid & 31;
  const int w    = tid >> 5;
  const int wm   = w & 3;    // row tile 0..3 (16 rows each)
  const int wn   = w >> 2;   // column half 0..1 (64 cols each)
  const int m0 = blockIdx.y * 64;
  const int n0 = blockIdx.x * 128;

  v8f acc[4] = {};

  const int ar = tid >> 2, ac = (tid & 3) * 8;   // A stage: 64 rows x 32 K
  const int bc  = tid >> 1;                      // B stage: 128 cols ...
  const int bk0 = (tid & 1) * 16;                // ... x 32 K (16 halves/thread)
  const int gr = m0 + ar;
  const int sr = (flags & FLAG_REMAP) ? ((gr >> 11) * 2565 + (gr & 2047) + 1) : gr;
  const int r  = lane & 15;
  const int hb = (lane >> 4) * 16;               // lane-half byte offset in K-row

  for (int k0 = 0; k0 < K; k0 += 32) {
    const _Float16* ap = A + (size_t)sr * K + k0 + ac;
    const _Float16* bp =
        Bp + (((size_t)(k0 >> 5) * N + (n0 + bc)) << 5) + bk0;
#if RQ_HAVE_ASYNC
    if (gr < M)
      async_b128(ap, &As[ar][ac]);
    else
      *reinterpret_cast<float4*>(&As[ar][ac]) = make_float4(0.f, 0.f, 0.f, 0.f);
    async_b128(bp,     &Bs[bc][bk0]);
    async_b128(bp + 8, &Bs[bc][bk0 + 8]);
#else
    if (gr < M)
      *reinterpret_cast<float4*>(&As[ar][ac]) = *reinterpret_cast<const float4*>(ap);
    else
      *reinterpret_cast<float4*>(&As[ar][ac]) = make_float4(0.f, 0.f, 0.f, 0.f);
    *reinterpret_cast<float4*>(&Bs[bc][bk0])     = *reinterpret_cast<const float4*>(bp);
    *reinterpret_cast<float4*>(&Bs[bc][bk0 + 8]) = *reinterpret_cast<const float4*>(bp + 8);
#endif

    if (k0 + 32 < K) {                           // global_prefetch_b8 next K tile
      if (gr < M) __builtin_prefetch((const void*)(ap + 32), 0, 0);
      __builtin_prefetch((const void*)(bp + ((size_t)N << 5)), 0, 0);
    }
#if RQ_HAVE_ASYNC
    wait_async0();
#endif
    __syncthreads();

    // Preload A + all 4 B fragments, then issue the 4 WMMAs back-to-back
    // (independent accumulators -> no D->A/B hazards, single DS drain).
    FragU af;
    const char* arow = (const char*)&As[wm * 16 + r][0];
    af.q[0] = *reinterpret_cast<const uint4*>(arow + hb);
    af.q[1] = *reinterpret_cast<const uint4*>(arow + hb + 32);
    FragU bf[4];
#pragma unroll
    for (int t = 0; t < 4; ++t) {
      const char* bcol = (const char*)&Bs[wn * 64 + t * 16 + r][0];
      bf[t].q[0] = *reinterpret_cast<const uint4*>(bcol + hb);
      bf[t].q[1] = *reinterpret_cast<const uint4*>(bcol + hb + 32);
    }
#pragma unroll
    for (int t = 0; t < 4; ++t)
      acc[t] = __builtin_amdgcn_wmma_f32_16x16x32_f16(false, af.v, false, bf[t].v,
                                                      (short)0, acc[t], false, false);
    __syncthreads();
  }

  const int mrow0 = m0 + wm * 16 + (lane >> 4) * 8;
#pragma unroll
  for (int t = 0; t < 4; ++t) {
    const int col = n0 + wn * 64 + t * 16 + r;
    const float bi = bias ? bias[col] : 0.f;
#pragma unroll
    for (int i = 0; i < 8; ++i) {
      const int row = mrow0 + i;
      if (row >= M) continue;
      float v = acc[t][i] + bi;
      if (flags & FLAG_GELU) v = gelu_exact(v);
      if (residual) v += residual[(size_t)row * N + col];
      if (Cf) Cf[(size_t)row * N + col] = v;
      if (Ch) Ch[(size_t)row * N + col] = (_Float16)v;
    }
  }
}

// ---------------------------------------------------------------------------
// Causal attention, online softmax. One wave per (seq, head, query row);
// lane owns 2 of the 64 head dims. qkv rows: [seq*N + i][1536] (q|k|v).
__global__ __launch_bounds__(256) void attn_kernel(
    const float* __restrict__ qkv, float* __restrict__ o, int N) {
  const int w    = threadIdx.x >> 5;
  const int lane = threadIdx.x & 31;
  const int i = blockIdx.x * 8 + w;
  if (i >= N) return;
  const int h   = blockIdx.y;
  const int seq = blockIdx.z;
  const float* qp = qkv + (size_t)(seq * N + i) * (3 * INNER) + h * DHEAD;
  const float scale = 0.125f;  // DHEAD^-0.5
  const float qa = qp[2 * lane] * scale;
  const float qb = qp[2 * lane + 1] * scale;
  float m = -1e30f, den = 0.f, a0 = 0.f, a1 = 0.f;
  for (int j = 0; j <= i; ++j) {
    const float* kp = qkv + (size_t)(seq * N + j) * (3 * INNER) + INNER + h * DHEAD;
    float p = qa * kp[2 * lane] + qb * kp[2 * lane + 1];
    p += __shfl_xor(p, 16);
    p += __shfl_xor(p, 8);
    p += __shfl_xor(p, 4);
    p += __shfl_xor(p, 2);
    p += __shfl_xor(p, 1);
    const float mn   = fmaxf(m, p);
    const float corr = __expf(m - mn);
    const float e    = __expf(p - mn);
    const float* vp  = kp + INNER;
    a0 = a0 * corr + e * vp[2 * lane];
    a1 = a1 * corr + e * vp[2 * lane + 1];
    den = den * corr + e;
    m = mn;
  }
  float* op = o + (size_t)(seq * N + i) * INNER + h * DHEAD;
  op[2 * lane]     = a0 / den;
  op[2 * lane + 1] = a1 / den;
}

// ---------------------------------------------------------------------------
// dt[(b*513+s)*5 + t][:] = t==0 ? spatial_final : (s<512 ? tokens[b,s,t-1] : 0)
__global__ __launch_bounds__(256) void build_dt(
    const float* __restrict__ spatial, const float* __restrict__ tokens,
    float* __restrict__ dt) {
  const int s = blockIdx.x;  // 0..512
  const int t = blockIdx.y;  // 0..4
  const int b = blockIdx.z;
  const size_t row = ((size_t)(b * NSP + s)) * NDP + t;
  float* drow = dt + row * DIMF;
  for (int c = threadIdx.x; c < DIMF; c += 256) {
    float v;
    if (t == 0)        v = spatial[((size_t)(b * NSP + s)) * DIMF + c];
    else if (s < MAXS) v = tokens[(((size_t)(b * MAXS + s) * DEPTHT) + (t - 1)) * DIMF + c];
    else               v = 0.f;
    drow[c] = v;
  }
}

// ---------------------------------------------------------------------------
extern "C" void kernel_launch(void* const* d_in, const int* in_sizes, int n_in,
                              void* d_out, int out_size, void* d_ws, size_t ws_size,
                              hipStream_t stream) {
  (void)in_sizes; (void)n_in; (void)out_size; (void)ws_size;

  const int*   ids       = (const int*)  d_in[0];
  const float* tok_emb   = (const float*)d_in[1];
  const float* start_tok = (const float*)d_in[2];
  const float* sp_pos    = (const float*)d_in[3];
  const float* dp_pos    = (const float*)d_in[4];
  const float* w_logit   = (const float*)d_in[5];
  const float* b_logit   = (const float*)d_in[6];
  const float* sp_ln1g = (const float*)d_in[7];
  const float* sp_ln1b = (const float*)d_in[8];
  const float* sp_qkv  = (const float*)d_in[9];
  const float* sp_wo   = (const float*)d_in[10];
  const float* sp_ln2g = (const float*)d_in[11];
  const float* sp_ln2b = (const float*)d_in[12];
  const float* sp_w1   = (const float*)d_in[13];
  const float* sp_b1   = (const float*)d_in[14];
  const float* sp_w2   = (const float*)d_in[15];
  const float* sp_b2   = (const float*)d_in[16];
  const float* sp_lnfg = (const float*)d_in[17];
  const float* sp_lnfb = (const float*)d_in[18];
  const float* dp_ln1g = (const float*)d_in[19];
  const float* dp_ln1b = (const float*)d_in[20];
  const float* dp_qkv  = (const float*)d_in[21];
  const float* dp_wo   = (const float*)d_in[22];
  const float* dp_ln2g = (const float*)d_in[23];
  const float* dp_ln2b = (const float*)d_in[24];
  const float* dp_w1   = (const float*)d_in[25];
  const float* dp_b1   = (const float*)d_in[26];
  const float* dp_w2   = (const float*)d_in[27];
  const float* dp_b2   = (const float*)d_in[28];
  const float* dp_lnfg = (const float*)d_in[29];
  const float* dp_lnfb = (const float*)d_in[30];

  // ---- workspace carve ----
  char*  base = (char*)d_ws;
  size_t off  = 0;
  auto ALLOC = [&](size_t bytes) -> char* {
    char* p = base + off;
    off = (off + bytes + 255) & ~(size_t)255;
    return p;
  };
  _Float16* wlogit_p = (_Float16*)ALLOC((size_t)DIMF * VOCAB * 2);
  _Float16* spqkv_p  = (_Float16*)ALLOC((size_t)4 * DIMF * 3 * INNER * 2);
  _Float16* spwo_p   = (_Float16*)ALLOC((size_t)4 * INNER * DIMF * 2);
  _Float16* spw1_p   = (_Float16*)ALLOC((size_t)4 * DIMF * 4 * DIMF * 2);
  _Float16* spw2_p   = (_Float16*)ALLOC((size_t)4 * 4 * DIMF * DIMF * 2);
  _Float16* dpqkv_p  = (_Float16*)ALLOC((size_t)2 * DIMF * 3 * INNER * 2);
  _Float16* dpwo_p   = (_Float16*)ALLOC((size_t)2 * INNER * DIMF * 2);
  _Float16* dpw1_p   = (_Float16*)ALLOC((size_t)2 * DIMF * 4 * DIMF * 2);
  _Float16* dpw2_p   = (_Float16*)ALLOC((size_t)2 * 4 * DIMF * DIMF * 2);
  float*    tokens_f = (float*)ALLOC((size_t)BATCH * MAXS * DEPTHT * DIMF * 4);
  float*    xsp      = (float*)ALLOC((size_t)MSP * DIMF * 4);
  float*    xdp      = (float*)ALLOC((size_t)MDP * DIMF * 4);
  _Float16* xn_h     = (_Float16*)ALLOC((size_t)MDP * DIMF * 2);
  float*    qkv_f    = (float*)ALLOC((size_t)MDP * 3 * INNER * 4);
  float*    o_f      = (float*)ALLOC((size_t)MDP * DIMF * 4);
  _Float16* h1_h     = (_Float16*)ALLOC((size_t)MDP * 4 * DIMF * 2);

  auto pack = [&](const float* s, _Float16* d, size_t KN, int N) {
    pack_w<<<2048, 256, 0, stream>>>(s, d, KN, N);
  };
  pack(w_logit, wlogit_p, (size_t)DIMF * VOCAB, VOCAB);
  pack(sp_qkv, spqkv_p, (size_t)4 * DIMF * 3 * INNER, 3 * INNER);
  pack(sp_wo,  spwo_p,  (size_t)4 * INNER * DIMF, DIMF);
  pack(sp_w1,  spw1_p,  (size_t)4 * DIMF * 4 * DIMF, 4 * DIMF);
  pack(sp_w2,  spw2_p,  (size_t)4 * 4 * DIMF * DIMF, DIMF);
  pack(dp_qkv, dpqkv_p, (size_t)2 * DIMF * 3 * INNER, 3 * INNER);
  pack(dp_wo,  dpwo_p,  (size_t)2 * INNER * DIMF, DIMF);
  pack(dp_w1,  dpw1_p,  (size_t)2 * DIMF * 4 * DIMF, 4 * DIMF);
  pack(dp_w2,  dpw2_p,  (size_t)2 * 4 * DIMF * DIMF, DIMF);

  embed_kernel<<<dim3(NSP, BATCH), 256, 0, stream>>>(
      ids, tok_emb, start_tok, sp_pos, dp_pos, tokens_f, xsp);

  auto gemm = [&](const _Float16* A, const _Float16* B, const float* bias,
                  const float* res, float* Cf, _Float16* Ch,
                  int M, int N, int K, int flags) {
    dim3 g(N / 128, (M + 63) / 64);
    gemm_wmma_f16<<<g, 256, 0, stream>>>(A, B, bias, res, Cf, Ch, M, N, K, flags);
  };
  auto cvt = [&](const float* s, _Float16* d, size_t n) {
    cvt_f32_to_f16<<<1024, 256, 0, stream>>>(s, d, n);
  };

  // ---- spatial transformer (4 layers, 1026 rows) ----
  for (int L = 0; L < 4; ++L) {
    ln512<<<MSP, 128, 0, stream>>>(xsp, sp_ln1g + L * DIMF, sp_ln1b + L * DIMF, xn_h, nullptr);
    gemm(xn_h, spqkv_p + (size_t)L * DIMF * 3 * INNER, nullptr, nullptr,
         qkv_f, nullptr, MSP, 3 * INNER, DIMF, 0);
    attn_kernel<<<dim3((NSP + 7) / 8, HEADS, BATCH), 256, 0, stream>>>(qkv_f, o_f, NSP);
    cvt(o_f, xn_h, (size_t)MSP * DIMF);
    gemm(xn_h, spwo_p + (size_t)L * INNER * DIMF, nullptr, xsp,
         xsp, nullptr, MSP, DIMF, INNER, 0);
    ln512<<<MSP, 128, 0, stream>>>(xsp, sp_ln2g + L * DIMF, sp_ln2b + L * DIMF, xn_h, nullptr);
    gemm(xn_h, spw1_p + (size_t)L * DIMF * 4 * DIMF, sp_b1 + (size_t)L * 4 * DIMF, nullptr,
         nullptr, h1_h, MSP, 4 * DIMF, DIMF, FLAG_GELU);
    gemm(h1_h, spw2_p + (size_t)L * 4 * DIMF * DIMF, sp_b2 + L * DIMF, xsp,
         xsp, nullptr, MSP, DIMF, 4 * DIMF, 0);
  }
  ln512<<<MSP, 128, 0, stream>>>(xsp, sp_lnfg, sp_lnfb, nullptr, o_f);
  build_dt<<<dim3(NSP, NDP, BATCH), 256, 0, stream>>>(o_f, tokens_f, xdp);

  // ---- depth transformer (2 layers, 5130 rows, seq len 5) ----
  for (int L = 0; L < 2; ++L) {
    ln512<<<MDP, 128, 0, stream>>>(xdp, dp_ln1g + L * DIMF, dp_ln1b + L * DIMF, xn_h, nullptr);
    gemm(xn_h, dpqkv_p + (size_t)L * DIMF * 3 * INNER, nullptr, nullptr,
         qkv_f, nullptr, MDP, 3 * INNER, DIMF, 0);
    attn_kernel<<<dim3(1, HEADS, MSP), 256, 0, stream>>>(qkv_f, o_f, NDP);
    cvt(o_f, xn_h, (size_t)MDP * DIMF);
    gemm(xn_h, dpwo_p + (size_t)L * INNER * DIMF, nullptr, xdp,
         xdp, nullptr, MDP, DIMF, INNER, 0);
    ln512<<<MDP, 128, 0, stream>>>(xdp, dp_ln2g + L * DIMF, dp_ln2b + L * DIMF, xn_h, nullptr);
    gemm(xn_h, dpw1_p + (size_t)L * DIMF * 4 * DIMF, dp_b1 + (size_t)L * 4 * DIMF, nullptr,
         nullptr, h1_h, MDP, 4 * DIMF, DIMF, FLAG_GELU);
    gemm(h1_h, dpw2_p + (size_t)L * 4 * DIMF * DIMF, dp_b2 + L * DIMF, xdp,
         xdp, nullptr, MDP, DIMF, 4 * DIMF, 0);
  }
  ln512<<<MDP, 128, 0, stream>>>(xdp, dp_lnfg, dp_lnfb, xn_h, nullptr);

  // ---- logits: only the 4096 rows that survive the slice, straight to d_out ----
  gemm(xn_h, wlogit_p, b_logit, nullptr, (float*)d_out, nullptr,
       BATCH * 2048, VOCAB, DIMF, FLAG_REMAP);
}
